// Model_29798483099751
// MI455X (gfx1250) — matrix operand order repaired and checked by reading, not compiled, not run on hardware
//
#include <hip/hip_runtime.h>
#include <hip/hip_bf16.h>

// ---------------- problem constants (from reference) ----------------
#define HDIM      768
#define K2H       1536          // 2*H
#define A_DIM     768
#define N_TREE    40000
#define N_CODES   10000
#define MAX_ANC   8
#define BB        16
#define VV        32
#define MM        48
#define OUTDIM    512
#define VERY_NEG  (-1e30f)

// WMMA tiling
#define KSTEP     32            // K per v_wmma_f32_16x16x32_f16
#define NTILES    (A_DIM / 16)  // 48 column tiles
#define ROWS_BLK  64            // 64 rows = 8 codes per workgroup
#define CODES_BLK 8
#define RGROUPS   4             // 4 row-groups of 16 rows
#define WAVES     8             // 256 threads
#define CT_PER_WV (NTILES / WAVES) // 6 col tiles per wave
#define QK        384           // K-quarter staged in LDS per phase

typedef _Float16 v8h  __attribute__((ext_vector_type(8)));
typedef _Float16 v16h __attribute__((ext_vector_type(16)));
typedef float    v8f  __attribute__((ext_vector_type(8)));

union Frag16 { v16h v; v8h h[2]; };

// workspace layout
#define W1P_ELEMS ((K2H / KSTEP) * NTILES * 32 * 16)   // 1,179,648
#define W1P_BYTES (W1P_ELEMS * 2)                      // 2,359,296
#define DAG_BYTES ((size_t)N_CODES * HDIM * 4)         // 30,720,000
#define EH_BYTES  ((size_t)N_TREE * HDIM * 2)          // 61,440,000
#define WS_NEED_F16TAB (W1P_BYTES + DAG_BYTES + EH_BYTES)

// --------------------------------------------------------------------
// Kernel 0: convert w1 (fp32, row-major K2H x A_DIM) into f16 WMMA
// B-fragment layout: lane L of each (kb, ct) tile holds 16 contiguous
// f16 = w1[kb + (L>>4)*16 + j, ct*16 + (L&15)], j = 0..15.
// --------------------------------------------------------------------
__global__ void pack_w1_kernel(const float* __restrict__ w1,
                               _Float16* __restrict__ w1p) {
    int idx = blockIdx.x * 256 + threadIdx.x;
    if (idx >= W1P_ELEMS) return;
    int j     = idx & 15;
    int rest  = idx >> 4;
    int lane  = rest & 31;
    int rest2 = rest >> 5;
    int ct    = rest2 % NTILES;
    int kbi   = rest2 / NTILES;
    int k     = kbi * KSTEP + (lane >> 4) * 16 + j;
    int col   = ct * 16 + (lane & 15);
    w1p[idx] = (_Float16)w1[(size_t)k * A_DIM + col];
}

// --------------------------------------------------------------------
// Kernel 0b: one-time fp32 -> f16 conversion of the embed table.
// --------------------------------------------------------------------
__global__ void cvt_embed_kernel(const float* __restrict__ embed,
                                 _Float16* __restrict__ eh) {
    size_t i = ((size_t)blockIdx.x * 256 + threadIdx.x) * 4;
    if (i >= (size_t)N_TREE * HDIM) return;
    const float4 v = *(const float4*)&embed[i];
    _Float16* d = &eh[i];
    d[0] = (_Float16)v.x;
    d[1] = (_Float16)v.y;
    d[2] = (_Float16)v.z;
    d[3] = (_Float16)v.w;
}

// --------------------------------------------------------------------
// Kernel 1: fused  X@w1 -> relu -> @w2 -> masked softmax -> dag_emb
// One block = 8 codes (64 rows). 256 threads = 8 waves.
// K=1536 processed in 4 staged quarters (leaf-lo/leaf-hi/anc-lo/anc-hi,
// 64 rows x 384 f16 = 48 KB LDS each); accumulators persist across
// phases. Each B fragment feeds 4 row-group WMMAs and is software-
// pipelined one tile ahead.
// --------------------------------------------------------------------
__global__ void __launch_bounds__(256)
attn_dag_kernel(const float* __restrict__ embed,
                const _Float16* __restrict__ eh,      // may be null
                const float* __restrict__ b1,
                const float* __restrict__ w2,
                const float* __restrict__ b2,
                const float* __restrict__ masks,
                const int*   __restrict__ leaves,
                const int*   __restrict__ ancestors,
                const _Float16* __restrict__ w1p,
                float* __restrict__ dag) {
    __shared__ _Float16 ldsA[ROWS_BLK * QK];     // one K-quarter of masked rows (48 KB)
    __shared__ float s_w2[A_DIM];
    __shared__ float s_b1[A_DIM];
    __shared__ float s_score[ROWS_BLK];
    __shared__ float s_attn[ROWS_BLK];           // softmax weight * mask
    __shared__ float s_mask[ROWS_BLK];
    __shared__ int   s_leaf[ROWS_BLK];
    __shared__ int   s_anc[ROWS_BLK];

    const int tid  = threadIdx.x;
    const int blk  = blockIdx.x;            // 0..1249
    const int row0 = blk * ROWS_BLK;        // global row = code*8 + anc

    if (tid < ROWS_BLK) {
        s_leaf[tid]  = leaves[row0 + tid];
        s_anc[tid]   = ancestors[row0 + tid];
        s_mask[tid]  = masks[row0 + tid];
        s_score[tid] = 0.0f;
    }
    for (int i = tid; i < A_DIM; i += 256) {
        s_w2[i] = w2[i];
        s_b1[i] = b1[i];
    }

    const int wave = tid >> 5;
    const int lane = tid & 31;
    const int n    = lane & 15;
    const int hi   = lane >> 4;
    const int ctb  = wave * CT_PER_WV;

    v8f acc[RGROUPS][CT_PER_WV];
    #pragma unroll
    for (int rg = 0; rg < RGROUPS; ++rg)
        #pragma unroll
        for (int c = 0; c < CT_PER_WV; ++c) acc[rg][c] = (v8f)(0.0f);

    const _Float16* ar[RGROUPS];
    #pragma unroll
    for (int rg = 0; rg < RGROUPS; ++rg)
        ar[rg] = &ldsA[(rg * 16 + (lane & 15)) * QK];

    // phases: p>>1 = 0 leaf / 1 ancestor;  p&1 = low/high K-quarter
    for (int p = 0; p < 4; ++p) {
        __syncthreads();   // stage buffer free / ids ready

        // ---- stage 64 rows of this K-quarter (masked, f16) ----
        if (eh) {
            // f16 table path: b128 load + binary-mask select (masks are 0/1)
            for (int idx = tid; idx < ROWS_BLK * (QK / 8); idx += 256) {
                int row = idx / (QK / 8);
                int c8  = (idx % (QK / 8)) * 8;
                int id  = (p & 2) ? s_anc[row] : s_leaf[row];
                int col = (p & 1) * QK + c8;
                v8h v = *(const v8h*)&eh[(size_t)id * HDIM + col];
                v8h z = {};
                if (s_mask[row] == 0.0f) v = z;
                *(v8h*)&ldsA[row * QK + c8] = v;
            }
        } else {
            // fp32 fallback path
            for (int idx = tid; idx < ROWS_BLK * (QK / 4); idx += 256) {
                int row = idx / (QK / 4);
                int c4  = (idx % (QK / 4)) * 4;
                int id  = (p & 2) ? s_anc[row] : s_leaf[row];
                int col = (p & 1) * QK + c4;
                const float4 v = *(const float4*)&embed[(size_t)id * HDIM + col];
                float mk = s_mask[row];
                _Float16* dst = &ldsA[row * QK + c4];
                dst[0] = (_Float16)(v.x * mk);
                dst[1] = (_Float16)(v.y * mk);
                dst[2] = (_Float16)(v.z * mk);
                dst[3] = (_Float16)(v.w * mk);
            }
        }
        __syncthreads();

        // ---- WMMA over this K-quarter: each B fragment feeds 4 row-groups,
        //      B software-pipelined one col-tile ahead ----
        for (int kb = 0; kb < QK; kb += KSTEP) {
            Frag16 a[RGROUPS];
            #pragma unroll
            for (int rg = 0; rg < RGROUPS; ++rg) {
                a[rg].h[0] = *(const v8h*)&ar[rg][kb + hi * 16];
                a[rg].h[1] = *(const v8h*)&ar[rg][kb + hi * 16 + 8];
            }

            int kbi = (p >> 1) * (HDIM / KSTEP) + (p & 1) * (QK / KSTEP) + (kb >> 5);
            const _Float16* bbase =
                w1p + (((size_t)kbi * NTILES + ctb) * 32 + lane) * 16;
            if (kb + KSTEP < QK)
                __builtin_prefetch(bbase + (size_t)NTILES * 32 * 16, 0, 1);

            Frag16 bcur;
            bcur.h[0] = *(const v8h*)bbase;
            bcur.h[1] = *(const v8h*)(bbase + 8);

            #pragma unroll
            for (int c = 0; c < CT_PER_WV; ++c) {
                Frag16 bnxt;
                if (c + 1 < CT_PER_WV) {
                    const _Float16* bp = bbase + (size_t)(c + 1) * 32 * 16;
                    bnxt.h[0] = *(const v8h*)bp;
                    bnxt.h[1] = *(const v8h*)(bp + 8);
                }
                #pragma unroll
                for (int rg = 0; rg < RGROUPS; ++rg)
                    acc[rg][c] = __builtin_amdgcn_wmma_f32_16x16x32_f16(
                        false, a[rg].v, false, bcur.v, (short)0, acc[rg][c], false, false);
                if (c + 1 < CT_PER_WV) bcur = bnxt;
            }
        }
    }

    // ---- relu(acc + b1) dot w2 -> per-row score (LDS float atomics) ----
    #pragma unroll
    for (int rg = 0; rg < RGROUPS; ++rg) {
        #pragma unroll
        for (int j = 0; j < 8; ++j) {
            float s = 0.0f;
            #pragma unroll
            for (int c = 0; c < CT_PER_WV; ++c) {
                int d = (ctb + c) * 16 + n;
                float h = acc[rg][c][j] + s_b1[d];
                h = fmaxf(h, 0.0f);
                s += h * s_w2[d];
            }
            atomicAdd(&s_score[rg * 16 + j + 8 * hi], s);
        }
    }
    __syncthreads();

    // ---- masked softmax over the 8 ancestors of each code ----
    if (tid < CODES_BLK) {
        float b2v = b2[0];
        float sc[MAX_ANC];
        float mx = -3.0e38f;
        #pragma unroll
        for (int a2 = 0; a2 < MAX_ANC; ++a2) {
            float mval = s_mask[tid * MAX_ANC + a2];
            float x = s_score[tid * MAX_ANC + a2] + b2v + (1.0f - mval) * VERY_NEG;
            sc[a2] = x;
            mx = fmaxf(mx, x);
        }
        float den = 0.0f;
        #pragma unroll
        for (int a2 = 0; a2 < MAX_ANC; ++a2) {
            sc[a2] = __expf(sc[a2] - mx);
            den += sc[a2];
        }
        float inv = 1.0f / den;
        #pragma unroll
        for (int a2 = 0; a2 < MAX_ANC; ++a2)
            s_attn[tid * MAX_ANC + a2] =
                sc[a2] * inv * s_mask[tid * MAX_ANC + a2];  // fold mask in
    }
    __syncthreads();

    // ---- dag_emb = sum_a (attn*m) * ancestors  (fp32 gather, L2-resident) ----
    for (int idx = tid; idx < CODES_BLK * HDIM; idx += 256) {
        int code = idx / HDIM;
        int h    = idx % HDIM;
        float a2sum = 0.0f;
        #pragma unroll
        for (int a2 = 0; a2 < MAX_ANC; ++a2) {
            int r = code * MAX_ANC + a2;
            a2sum += s_attn[r] * embed[(size_t)s_anc[r] * HDIM + h];
        }
        dag[((size_t)blk * CODES_BLK + code) * HDIM + h] = a2sum;
    }
}

// --------------------------------------------------------------------
// Kernel 2: gather dag_emb by input_ids, masked mean over M, @ wc + bc
// One block per (b,v).   (0.4 GFLOP total -> VALU is plenty.)
// --------------------------------------------------------------------
__global__ void __launch_bounds__(256)
pool_out_kernel(const float* __restrict__ dag,
                const float* __restrict__ wc,
                const float* __restrict__ bc,
                const float* __restrict__ code_mask,
                const int*   __restrict__ input_ids,
                float* __restrict__ out) {
    __shared__ float sp[HDIM];
    __shared__ int   sid[MM];
    __shared__ float scm[MM];
    __shared__ float sden;

    const int bv  = blockIdx.x;        // 0..511
    const int tid = threadIdx.x;

    if (tid < MM) {
        sid[tid] = input_ids[bv * MM + tid];
        scm[tid] = code_mask[bv * MM + tid];
    }
    __syncthreads();
    if (tid == 0) {
        float s = 0.0f;
        for (int m = 0; m < MM; ++m) s += scm[m];
        sden = 1.0f / fmaxf(s, 1.0f);
    }
    __syncthreads();

    for (int h = tid; h < HDIM; h += 256) {
        float a = 0.0f;
        for (int m = 0; m < MM; ++m) {
            int id  = sid[m];
            float c = scm[m];
            if (id > 0 && c != 0.0f)
                a += dag[(size_t)(id - 1) * HDIM + h] * c;
        }
        sp[h] = a * sden;
    }
    __syncthreads();

    for (int o = tid; o < OUTDIM; o += 256) {
        float a = bc[o];
        for (int h = 0; h < HDIM; ++h)
            a += sp[h] * wc[(size_t)h * OUTDIM + o];
        out[(size_t)bv * OUTDIM + o] = a;
    }
}

// --------------------------------------------------------------------
extern "C" void kernel_launch(void* const* d_in, const int* in_sizes, int n_in,
                              void* d_out, int out_size, void* d_ws, size_t ws_size,
                              hipStream_t stream) {
    const float* embed     = (const float*)d_in[0];
    const float* w1        = (const float*)d_in[1];
    const float* b1        = (const float*)d_in[2];
    const float* w2        = (const float*)d_in[3];
    const float* b2        = (const float*)d_in[4];
    const float* wc        = (const float*)d_in[5];
    const float* bc        = (const float*)d_in[6];
    const float* masks     = (const float*)d_in[7];
    const float* code_mask = (const float*)d_in[8];
    const int*   leaves    = (const int*)d_in[9];
    const int*   ancestors = (const int*)d_in[10];
    const int*   input_ids = (const int*)d_in[11];

    _Float16* w1p = (_Float16*)d_ws;
    float*    dag = (float*)((char*)d_ws + W1P_BYTES);
    float*    out = (float*)d_out;

    const bool use_f16tab = (ws_size >= WS_NEED_F16TAB);
    _Float16* eh = use_f16tab
        ? (_Float16*)((char*)d_ws + W1P_BYTES + DAG_BYTES)
        : (_Float16*)nullptr;

    pack_w1_kernel<<<(W1P_ELEMS + 255) / 256, 256, 0, stream>>>(w1, w1p);

    if (use_f16tab) {
        size_t quads = ((size_t)N_TREE * HDIM) / 4;           // 7,680,000
        cvt_embed_kernel<<<(int)((quads + 255) / 256), 256, 0, stream>>>(embed, eh);
    }

    attn_dag_kernel<<<N_CODES / CODES_BLK, 256, 0, stream>>>(
        embed, eh, b1, w2, b2, masks, leaves, ancestors, w1p, dag);

    pool_out_kernel<<<BB * VV, 256, 0, stream>>>(
        dag, wc, bc, code_mask, input_ids, out);
}
